// GATLayer_27513560498266
// MI455X (gfx1250) — matrix operand order repaired and checked by reading, not compiled, not run on hardware
//
#include <hip/hip_runtime.h>

// ---------------------------------------------------------------------------
// GAT layer for gfx1250 (MI455X): bf16 WMMA for both GEMMs, fused flash-style
// masked softmax-aggregation, fused skip+LayerNorm.
// ---------------------------------------------------------------------------

typedef __attribute__((ext_vector_type(8)))  float          v8f;
typedef __attribute__((ext_vector_type(16))) __bf16         v16bf;
typedef __attribute__((ext_vector_type(8)))  unsigned short ushort8;

union V16U {
    v16bf v;
    ushort8 h8[2];
    unsigned short u[16];
};

#define DEVI __device__ __forceinline__

static constexpr int B_   = 8;
static constexpr int N_   = 1024;
static constexpr int FIN  = 512;
static constexpr int H_   = 8;
static constexpr int FOUT = 64;
static constexpr int D_   = 512;   // H_*FOUT == FIN

DEVI unsigned short f2bf(float f) {
    unsigned u = __builtin_bit_cast(unsigned, f);
    unsigned r = u + 0x7FFFu + ((u >> 16) & 1u);   // RNE
    return (unsigned short)(r >> 16);
}
DEVI float bf2f(unsigned short h) {
    unsigned u = ((unsigned)h) << 16;
    return __builtin_bit_cast(float, u);
}

// ---------------------------------------------------------------------------
// Prep: cast x -> bf16 (row-major), W -> bf16 transposed (Wt[n][k])
// ---------------------------------------------------------------------------
__global__ void k_cast_x(const float* __restrict__ x, unsigned short* __restrict__ xb) {
    int i = blockIdx.x * 256 + threadIdx.x;          // B_*N_*FIN threads
    xb[i] = f2bf(x[i]);
}

__global__ void k_cast_wt(const float* __restrict__ W, unsigned short* __restrict__ wt) {
    int i = blockIdx.x * 256 + threadIdx.x;          // FIN*D_ threads
    int n = i >> 9;            // output feature
    int k = i & 511;           // input feature
    wt[i] = f2bf(W[k * D_ + n]);                     // wt[n*512 + k] = W[k][n]
}

// ---------------------------------------------------------------------------
// GEMM 1:  Wh = x @ W  (bf16 WMMA, f32 acc), output written TRANSPOSED as
//          vt[((b*H+h)*64 + f_local)*1024 + node]  (bf16)
// One wave per 16x64 output tile (one head-slab): A tile reused across 4
// column tiles -> 4 WMMAs per K-step, 10 b128 loads per 4 WMMAs.
// ---------------------------------------------------------------------------
__global__ void k_gemm(const unsigned short* __restrict__ xb,
                       const unsigned short* __restrict__ wt,
                       unsigned short* __restrict__ vt) {
    const int wid  = threadIdx.x >> 5;
    const int lane = threadIdx.x & 31;
    const int tile = blockIdx.x * 8 + wid;           // 512 * 8 = 4096 tiles
    const int m0 = (tile >> 3) * 16;                 // row tile (over B*N)
    const int n0 = (tile & 7) * 64;                  // head-slab (over D_)
    const int nlane = lane & 15;
    const int half  = lane >> 4;

    const unsigned short* arow = xb + (size_t)(m0 + nlane) * FIN;
    const unsigned short* brow = wt + (size_t)(n0 + nlane) * FIN;   // t-tiles at +t*16*FIN

    v8f acc[4] = {{}, {}, {}, {}};
    #pragma unroll
    for (int k0 = 0; k0 < FIN; k0 += 32) {
        V16U A;
        // A (16x32 bf16): elem i<8 -> K = half*8+i ; i>=8 -> K = 16+half*8+(i-8)
        A.h8[0] = *(const ushort8*)(arow + k0 + half * 8);
        A.h8[1] = *(const ushort8*)(arow + k0 + 16 + half * 8);
        #pragma unroll
        for (int t = 0; t < 4; ++t) {
            V16U Bm;
            // B (32x16 bf16): elem i<8 -> K = half*8+i ; i>=8 -> K = 16+half*8+(i-8)
            const unsigned short* bp = brow + t * 16 * FIN + k0 + half * 8;
            Bm.h8[0] = *(const ushort8*)(bp);
            Bm.h8[1] = *(const ushort8*)(bp + 16);
            acc[t] = __builtin_amdgcn_wmma_f32_16x16x32_bf16(
                false, A.v, false, Bm.v, (short)0, acc[t], false, false);
        }
    }

    // D layout: VGPR r -> row M = half*8 + r, col N = nlane
    const int b     = m0 >> 10;
    const int node0 = (m0 & 1023) + half * 8;        // 8 consecutive nodes
    const int h     = n0 >> 6;                       // whole slab is one head
    #pragma unroll
    for (int t = 0; t < 4; ++t) {
        const int fl = t * 16 + nlane;
        ushort8 st;
        #pragma unroll
        for (int r = 0; r < 8; ++r) st[r] = f2bf(acc[t][r]);
        *(ushort8*)(vt + (size_t)((b * H_ + h) * FOUT + fl) * N_ + node0) = st;
    }
}

// ---------------------------------------------------------------------------
// Edge scores: e_src/e_dst[b,h,node] = sum_f Wh[b,node,h,f] * a[h, f / 64+f]
// ---------------------------------------------------------------------------
__global__ void k_edge(const unsigned short* __restrict__ vt,
                       const float* __restrict__ a,
                       float* __restrict__ es, float* __restrict__ ed) {
    int idx  = blockIdx.x * 256 + threadIdx.x;       // B_*H_*N_ threads
    int node = idx & 1023;
    int bh   = idx >> 10;
    int h    = bh & 7;
    const unsigned short* vcol = vt + (size_t)bh * FOUT * N_ + node;
    const float* av = a + h * (2 * FOUT);
    float s0 = 0.f, s1 = 0.f;
    #pragma unroll 8
    for (int f = 0; f < FOUT; ++f) {
        float v = bf2f(vcol[f * N_]);
        s0 += v * av[f];
        s1 += v * av[FOUT + f];
    }
    es[idx] = s0;
    ed[idx] = s1;
}

// ---------------------------------------------------------------------------
// Softmax stats: per (b,h,i) row -> max and sum of exp over masked neighbors
// One wave32 per row, shuffle reductions.
// ---------------------------------------------------------------------------
__global__ void k_stats(const float* __restrict__ es, const float* __restrict__ ed,
                        const unsigned char* __restrict__ adj,
                        float* __restrict__ rm, float* __restrict__ rs) {
    const int wid  = threadIdx.x >> 5;
    const int lane = threadIdx.x & 31;
    const int gid  = blockIdx.x * 8 + wid;           // B_*H_*N_ waves
    const int i    = gid & 1023;
    const int bh   = gid >> 10;
    const int b    = bh >> 3;
    const float esr = es[bh * N_ + i];
    const float* edrow = ed + bh * N_;
    const unsigned char* adjrow = adj + (size_t)(b * N_ + i) * N_;

    float mx = -3.0e38f;
    for (int j = lane; j < N_; j += 32) {
        if (adjrow[j]) {
            float v = esr + edrow[j];
            v = (v >= 0.f) ? v : 0.2f * v;
            mx = fmaxf(mx, v);
        }
    }
    #pragma unroll
    for (int o = 16; o; o >>= 1) mx = fmaxf(mx, __shfl_xor(mx, o, 32));

    float s = 0.f;
    for (int j = lane; j < N_; j += 32) {
        if (adjrow[j]) {
            float v = esr + edrow[j];
            v = (v >= 0.f) ? v : 0.2f * v;
            s += __expf(v - mx);
        }
    }
    #pragma unroll
    for (int o = 16; o; o >>= 1) s += __shfl_xor(s, o, 32);

    if (lane == 0) { rm[bh * N_ + i] = mx; rs[bh * N_ + i] = s; }
}

// ---------------------------------------------------------------------------
// Aggregation:  h[b,i,h*64+f] = (1/l_i) * sum_j exp(e_ij - m_i) * Wh[b,j,h,f]
// One wave per (b,h, 16-row tile). Alpha 16x32 tile built in registers in the
// exact WMMA A-layout; 4 WMMAs per j-block (F_OUT = 4 x 16).
// ---------------------------------------------------------------------------
__global__ void k_aggr(const unsigned short* __restrict__ vt,
                       const float* __restrict__ es, const float* __restrict__ ed,
                       const float* __restrict__ rm, const float* __restrict__ rs,
                       const unsigned char* __restrict__ adj,
                       float* __restrict__ out) {
    const int wid  = threadIdx.x >> 5;
    const int lane = threadIdx.x & 31;
    const int gid  = blockIdx.x * 8 + wid;           // B_*H_*(N_/16) = 4096 waves
    const int i0   = (gid & 63) * 16;
    const int bh   = gid >> 6;
    const int h    = bh & 7;
    const int b    = bh >> 3;
    const int nlane = lane & 15;
    const int half  = lane >> 4;

    const int irow  = i0 + nlane;                    // A row handled by this lane
    const float esr  = es[bh * N_ + irow];
    const float mrow = rm[bh * N_ + irow];
    const float* edrow = ed + bh * N_;
    const unsigned char* adjrow = adj + (size_t)(b * N_ + irow) * N_;
    const unsigned short* vbase = vt + (size_t)bh * FOUT * N_;

    v8f acc[4] = {{}, {}, {}, {}};

    for (int j0 = 0; j0 < N_; j0 += 32) {
        // Build alpha-tile A (16x32 bf16) in register layout
        V16U A;
        #pragma unroll
        for (int i = 0; i < 16; ++i) {
            int kk = (i < 8) ? (half * 8 + i) : (16 + half * 8 + (i - 8));
            int jj = j0 + kk;
            float v = esr + edrow[jj];
            v = (v >= 0.f) ? v : 0.2f * v;
            float p = adjrow[jj] ? __expf(v - mrow) : 0.f;
            A.u[i] = f2bf(p);
        }
        const int jb = j0 + half * 8;
        #pragma unroll
        for (int t = 0; t < 4; ++t) {
            V16U Bm;
            const unsigned short* bp = vbase + (size_t)(t * 16 + nlane) * N_ + jb;
            Bm.h8[0] = *(const ushort8*)(bp);
            Bm.h8[1] = *(const ushort8*)(bp + 16);
            acc[t] = __builtin_amdgcn_wmma_f32_16x16x32_bf16(
                false, A.v, false, Bm.v, (short)0, acc[t], false, false);
        }
    }

    float linv[8];
    #pragma unroll
    for (int r = 0; r < 8; ++r)
        linv[r] = 1.0f / rs[bh * N_ + i0 + half * 8 + r];

    #pragma unroll
    for (int t = 0; t < 4; ++t) {
        #pragma unroll
        for (int r = 0; r < 8; ++r) {
            int gi = i0 + half * 8 + r;
            out[(size_t)(b * N_ + gi) * D_ + h * FOUT + t * 16 + nlane] = acc[t][r] * linv[r];
        }
    }
}

// ---------------------------------------------------------------------------
// Skip + LayerNorm, in place on d_out. One 256-thread block per (b,i) row.
// ---------------------------------------------------------------------------
__global__ void k_ln(float* __restrict__ out, const float* __restrict__ x,
                     const float* __restrict__ g, const float* __restrict__ bta) {
    const int row = blockIdx.x;
    const int t   = threadIdx.x;
    const size_t base = (size_t)row * D_;
    float v0 = out[base + t]       + x[base + t];
    float v1 = out[base + t + 256] + x[base + t + 256];
    float s = v0 + v1, q = v0 * v0 + v1 * v1;
    #pragma unroll
    for (int o = 16; o; o >>= 1) { s += __shfl_xor(s, o, 32); q += __shfl_xor(q, o, 32); }

    __shared__ float sh[16];
    const int wid = t >> 5, lane = t & 31;
    if (lane == 0) { sh[wid] = s; sh[8 + wid] = q; }
    __syncthreads();
    if (t == 0) {
        float S = 0.f, Q = 0.f;
        #pragma unroll
        for (int w = 0; w < 8; ++w) { S += sh[w]; Q += sh[8 + w]; }
        sh[0] = S; sh[8] = Q;
    }
    __syncthreads();
    const float mu   = sh[0] * (1.0f / D_);
    const float var  = sh[8] * (1.0f / D_) - mu * mu;
    const float rstd = rsqrtf(var + 1e-5f);
    out[base + t]       = (v0 - mu) * rstd * g[t]       + bta[t];
    out[base + t + 256] = (v1 - mu) * rstd * g[t + 256] + bta[t + 256];
}

// ---------------------------------------------------------------------------
// Host launch
// ---------------------------------------------------------------------------
extern "C" void kernel_launch(void* const* d_in, const int* in_sizes, int n_in,
                              void* d_out, int out_size, void* d_ws, size_t ws_size,
                              hipStream_t stream) {
    (void)in_sizes; (void)n_in; (void)out_size; (void)ws_size;
    const float*         x     = (const float*)d_in[0];
    const unsigned char* adj   = (const unsigned char*)d_in[1];   // jax bool -> 1 byte
    const float*         W     = (const float*)d_in[2];
    const float*         a     = (const float*)d_in[3];
    const float*         gamma = (const float*)d_in[4];
    const float*         beta  = (const float*)d_in[5];
    float*               out   = (float*)d_out;

    char* ws = (char*)d_ws;
    // Workspace layout (all offsets 256B aligned), ~17.5 MB total
    unsigned short* xb = (unsigned short*)(ws + 0);                     //  8 MB
    unsigned short* wt = (unsigned short*)(ws + 8388608);               // 512 KB
    unsigned short* vt = (unsigned short*)(ws + 8912896);               //  8 MB (Wh^T bf16)
    float*          es = (float*)(ws + 17301504);                       // 256 KB
    float*          ed = (float*)(ws + 17563648);                       // 256 KB
    float*          rm = (float*)(ws + 17825792);                       // 256 KB
    float*          rs = (float*)(ws + 18087936);                       // 256 KB

    // 1) casts
    k_cast_x <<<(B_ * N_ * FIN) / 256, 256, 0, stream>>>(x, xb);
    k_cast_wt<<<(FIN * D_) / 256,      256, 0, stream>>>(W, wt);

    // 2) GEMM: Wh = x @ W  (writes transposed bf16 vt)
    //    4096 tiles of 16x64, 8 waves/block -> 512 blocks
    k_gemm<<<512, 256, 0, stream>>>(xb, wt, vt);

    // 3) per-node attention scores
    k_edge<<<(B_ * H_ * N_) / 256, 256, 0, stream>>>(vt, a, es, ed);

    // 4) masked softmax stats (max, sumexp): 65536 waves, 8 waves/block
    k_stats<<<8192, 256, 0, stream>>>(es, ed, adj, rm, rs);

    // 5) flash-style aggregation GEMM: 4096 waves, 8 waves/block
    k_aggr<<<512, 256, 0, stream>>>(vt, es, ed, rm, rs, adj, out);

    // 6) skip + LayerNorm in place
    k_ln<<<B_ * N_, 256, 0, stream>>>(out, x, gamma, beta);
}
